// SymphoniesDecoder_53566832116244
// MI455X (gfx1250) — matrix operand order repaired and checked by reading, not compiled, not run on hardware
//
#include <hip/hip_runtime.h>
#include <hip/hip_bf16.h>
#include <math.h>

// ---------------- model / geometry constants ----------------
#define EMBED 128
#define HEADS 4
#define HDIM 32
#define NCLS 20
#define NLAYERS 2
#define SX 128
#define SY 128
#define SZ 16
#define NQ (SX * SY * SZ)   // 262144
#define NI 300
#define IMH 96
#define IMW 320
#define NPIX (IMH * IMW)    // 30720
#define NS_MAX 64           // cap on selected instances (expected ~8)
#define F_MAX 73728         // cap on fov voxels (expected ~65536 = 25% of NQ)
#define SCORE_THR 0.25f
#define MASK_THR 0.5f
#define IOU_THR 0.8f

typedef __attribute__((ext_vector_type(16))) _Float16 v16h;
typedef __attribute__((ext_vector_type(8)))  float    v8f;
typedef int b128_t __attribute__((__vector_size__(16)));  // matches async-LDS builtin pointee

__device__ __forceinline__ float sigmoidf_(float x) { return 1.f / (1.f + __expf(-x)); }

__device__ __forceinline__ v8f wmma_f16(v16h a, v16h b, v8f c) {
  // (neg_a, A, neg_b, B, c_mod, C, reuse_a, reuse_b)
  return __builtin_amdgcn_wmma_f32_16x16x32_f16(false, a, false, b, (short)0, c, false, false);
}

// A fragment (16xK row-major tile in LDS, f16).  ISA 7.12.2 16-bit A 16x32:
// lane = grp*16 + m ; element j -> k = j + (j>=8 ? 8 : 0) + grp*8
__device__ __forceinline__ v16h fragA_lds(const _Float16* a_sh, int K, int kbase, int lane) {
  const int m = lane & 15;
  const int grp = (lane >> 4) & 1;
  v16h f;
#pragma unroll
  for (int j = 0; j < 16; ++j) {
    const int kl = j + ((j & 8) ? 8 : 0) + grp * 8;
    f[j] = a_sh[m * K + kbase + kl];
  }
  return f;
}

// B fragment built from weight matrix W (N x K, row-major, f32): B[k][n] = W[n][k].
// Dense 16-bit B 32x16: element j -> k = j + (j>=8 ? 8 : 0) + grp*8 (same as A).
__device__ __forceinline__ v16h fragB_w(const float* W, int K, int Nvalid, int nbase,
                                        int kbase, int lane) {
  const int n = nbase + (lane & 15);
  const int grp = (lane >> 4) & 1;
  v16h f;
  if (n < Nvalid) {
#pragma unroll
    for (int j = 0; j < 16; ++j) {
      const int k = kbase + j + ((j & 8) ? 8 : 0) + grp * 8;
      f[j] = (_Float16)W[(size_t)n * K + k];
    }
  } else {
#pragma unroll
    for (int j = 0; j < 16; ++j) f[j] = (_Float16)0.f;
  }
  return f;
}

// ---------------- small / setup kernels ----------------

__global__ void k_zero_i32(int* p, int n) {
  int t = blockIdx.x * blockDim.x + threadIdx.x;
  if (t < n) p[t] = 0;
}
__global__ void k_zero_f32(float* p, int n) {
  int t = blockIdx.x * blockDim.x + threadIdx.x;
  if (t < n) p[t] = 0.f;
}

__global__ void k_invmats(const float* Km, const float* Em, float* invm) {
  // single thread: 3x3 adjugate inverse + 4x4 Gauss-Jordan inverse
  float a = Km[0], b = Km[1], c = Km[2], d = Km[3], e = Km[4], f = Km[5],
        g = Km[6], h = Km[7], i = Km[8];
  float det = a * (e * i - f * h) - b * (d * i - f * g) + c * (d * h - e * g);
  float id = 1.f / det;
  invm[0] = (e * i - f * h) * id; invm[1] = (c * h - b * i) * id; invm[2] = (b * f - c * e) * id;
  invm[3] = (f * g - d * i) * id; invm[4] = (a * i - c * g) * id; invm[5] = (c * d - a * f) * id;
  invm[6] = (d * h - e * g) * id; invm[7] = (b * g - a * h) * id; invm[8] = (a * e - b * d) * id;
  float M[4][8];
  for (int r = 0; r < 4; ++r)
    for (int cc = 0; cc < 4; ++cc) { M[r][cc] = Em[r * 4 + cc]; M[r][4 + cc] = (r == cc) ? 1.f : 0.f; }
  for (int col = 0; col < 4; ++col) {
    int piv = col; float best = fabsf(M[col][col]);
    for (int r = col + 1; r < 4; ++r)
      if (fabsf(M[r][col]) > best) { best = fabsf(M[r][col]); piv = r; }
    if (piv != col)
      for (int cc = 0; cc < 8; ++cc) { float t = M[col][cc]; M[col][cc] = M[piv][cc]; M[piv][cc] = t; }
    float ip = 1.f / M[col][col];
    for (int cc = 0; cc < 8; ++cc) M[col][cc] *= ip;
    for (int r = 0; r < 4; ++r)
      if (r != col) { float fc = M[r][col]; for (int cc = 0; cc < 8; ++cc) M[r][cc] -= fc * M[col][cc]; }
  }
  for (int r = 0; r < 4; ++r)
    for (int cc = 0; cc < 4; ++cc) invm[16 + r * 4 + cc] = M[r][4 + cc];
}

__global__ void k_scores(const float* __restrict__ logits, float* __restrict__ scores,
                         int* __restrict__ keepf) {
  int i = blockIdx.x * blockDim.x + threadIdx.x;
  if (i >= NI) return;
  float sg[NCLS + 1];
  float mx = -1e30f;
  for (int c = 0; c <= NCLS; ++c) {
    sg[c] = sigmoidf_(logits[i * (NCLS + 1) + c]) * (1.f / 0.06f);
    mx = fmaxf(mx, sg[c]);
  }
  float s = 0.f;
  for (int c = 0; c <= NCLS; ++c) { sg[c] = __expf(sg[c] - mx); s += sg[c]; }
  float best = -1.f; int bl = 0;
  for (int c = 0; c <= NCLS; ++c) { float p = sg[c] / s; if (p > best) { best = p; bl = c; } }
  scores[i] = best;
  keepf[i] = (bl != (NCLS + 1)) && (best > SCORE_THR);  // faithful: labels != shape[-1]
}

__global__ void k_compact_kidx(const int* keepf, int* kidx, int* cnts) {
  int n = 0;
  for (int i = 0; i < NI; ++i) if (keepf[i]) kidx[n++] = i;
  cnts[0] = n;
}

__global__ void k_mids(const float* __restrict__ pmasks, const float* __restrict__ scores,
                       const int* __restrict__ kidx, const int* __restrict__ cnts,
                       int* __restrict__ mids) {
  int p = blockIdx.x * blockDim.x + threadIdx.x;
  if (p >= NPIX) return;
  int nk = cnts[0];
  float best = -1e30f; int bi = 0;
  for (int t = 0; t < nk; ++t) {
    int inst = kidx[t];
    float v = scores[inst] * sigmoidf_(pmasks[(size_t)inst * NPIX + p]);
    if (v > best) { best = v; bi = t; }
  }
  mids[p] = bi;
}

__global__ void k_areas(const float* __restrict__ pmasks, const int* __restrict__ kidx,
                        const int* __restrict__ cnts, const int* __restrict__ mids,
                        int* marea, int* parea, int* matm, int* selmap) {
  int p = blockIdx.x * blockDim.x + threadIdx.x;
  if (p >= NPIX) return;
  int nk = cnts[0];
  int w = mids[p];
  int sm = -1;
  for (int t = 0; t < nk; ++t) {
    int pb = sigmoidf_(pmasks[(size_t)kidx[t] * NPIX + p]) >= MASK_THR;
    if (pb) atomicAdd(&parea[t], 1);
    if (t == w && nk > 0) {
      atomicAdd(&marea[t], 1);
      if (pb) { atomicAdd(&matm[t], 1); sm = t; }
    }
  }
  selmap[p] = sm;
}

__global__ void k_compact_sel(const int* cnts_in, const int* kidx, const int* marea,
                              const int* parea, const int* matm, int* remap, int* k2idx,
                              int* cnts) {
  int nk = cnts_in[0];
  int n = 0;
  for (int t = 0; t < nk; ++t) {
    int ok = (matm[t] > 0) && ((float)marea[t] >= IOU_THR * (float)parea[t]);
    if (ok && n < NS_MAX) { remap[t] = n; k2idx[n] = kidx[t]; ++n; }
    else remap[t] = -1;
  }
  cnts[1] = n;
}

__global__ void k_world(const float* __restrict__ depth, const float* __restrict__ invm,
                        float* __restrict__ world) {
  int p = blockIdx.x * blockDim.x + threadIdx.x;
  if (p >= NPIX) return;
  float gx = (float)(p % IMW) * 4.0f;   // 1280/320
  float gy = (float)(p / IMW) * 4.0f;   // 384/96
  float dd = depth[p];
  float v0 = gx * dd, v1 = gy * dd, v2 = dd;
  float c0 = invm[0] * v0 + invm[1] * v1 + invm[2] * v2;
  float c1 = invm[3] * v0 + invm[4] * v1 + invm[5] * v2;
  float c2 = invm[6] * v0 + invm[7] * v1 + invm[8] * v2;
  const float* ie = invm + 16;
  for (int r = 0; r < 3; ++r)
    world[(size_t)p * 3 + r] = ie[r * 4 + 0] * c0 + ie[r * 4 + 1] * c1 + ie[r * 4 + 2] * c2 + ie[r * 4 + 3];
}

__global__ void k_xyz(const int* __restrict__ selmap, const int* __restrict__ remap,
                      const float* __restrict__ world, float* __restrict__ xyz) {
  int p = blockIdx.x * blockDim.x + threadIdx.x;
  if (p >= NPIX) return;
  int sm = selmap[p];
  if (sm < 0) return;
  int j = remap[sm];
  if (j < 0) return;
  for (int r = 0; r < 3; ++r) atomicAdd(&xyz[j * 3 + r], world[(size_t)p * 3 + r]);
}

__global__ void k_instqp(const float* __restrict__ queries, const float* __restrict__ ipw,
                         const float* __restrict__ posw, const int* __restrict__ k2idx,
                         const float* __restrict__ xyz, const int* __restrict__ cnts,
                         float* __restrict__ instq, float* __restrict__ instp) {
  int t = blockIdx.x * blockDim.x + threadIdx.x;
  int j = t >> 7, c = t & 127;
  if (j >= cnts[1]) return;
  float mp = xyz[j * 3 + 0] * posw[c * 3 + 0] + xyz[j * 3 + 1] * posw[c * 3 + 1] +
             xyz[j * 3 + 2] * posw[c * 3 + 2];
  int src = k2idx[j];
  instp[(size_t)j * EMBED + c] = ipw[(size_t)src * EMBED + c] + mp;
  instq[(size_t)j * EMBED + c] = queries[(size_t)src * EMBED + c];
}

// ---------------- fov compaction (ordered) ----------------

__global__ void k_fov_count(const unsigned char* __restrict__ fm, int* __restrict__ bc) {
  __shared__ int sh[256];
  int q = blockIdx.x * 256 + threadIdx.x;
  sh[threadIdx.x] = fm[q] ? 1 : 0;
  __syncthreads();
  for (int st = 128; st > 0; st >>= 1) {
    if (threadIdx.x < st) sh[threadIdx.x] += sh[threadIdx.x + st];
    __syncthreads();
  }
  if (threadIdx.x == 0) bc[blockIdx.x] = sh[0];
}

__global__ void k_fov_scan(const int* __restrict__ bc, int* __restrict__ boff, int* cnts) {
  __shared__ int sh[1024];
  int t = threadIdx.x;
  int my = bc[t];
  sh[t] = my;
  __syncthreads();
  for (int st = 1; st < 1024; st <<= 1) {
    int v = (t >= st) ? sh[t - st] : 0;
    __syncthreads();
    sh[t] += v;
    __syncthreads();
  }
  boff[t] = sh[t] - my;   // exclusive prefix
  if (t == 1023) { int F = sh[1023]; cnts[2] = (F < F_MAX) ? F : F_MAX; }
}

__global__ void k_fov_scatter(const unsigned char* __restrict__ fm, const int* __restrict__ boff,
                              int* __restrict__ fidx) {
  __shared__ int sh[256];
  int q = blockIdx.x * 256 + threadIdx.x;
  int fl = fm[q] ? 1 : 0;
  sh[threadIdx.x] = fl;
  __syncthreads();
  for (int st = 1; st < 256; st <<= 1) {
    int v = (threadIdx.x >= st) ? sh[threadIdx.x - st] : 0;
    __syncthreads();
    sh[threadIdx.x] += v;
    __syncthreads();
  }
  int pos = boff[blockIdx.x] + sh[threadIdx.x] - fl;
  if (fl && pos < F_MAX) fidx[pos] = q;
}

__global__ void k_se_sp(const float* __restrict__ sew, const float* __restrict__ x3d,
                        const float* __restrict__ posw, const float* __restrict__ vorigin,
                        const int* __restrict__ fidx, const int* __restrict__ cnts,
                        float* __restrict__ se, float* __restrict__ sp) {
  int t = blockIdx.x * blockDim.x + threadIdx.x;
  int f = t >> 7, c = t & 127;
  if (f >= cnts[2]) return;
  int q = fidx[f];
  se[(size_t)f * EMBED + c] = sew[(size_t)q * EMBED + c] + x3d[(size_t)c * NQ + q];
  int ix = q >> 11;           // q / (SY*SZ)
  int rem = q & 2047;
  int iy = rem >> 4, iz = rem & 15;
  float px = (ix + 0.5f) * 0.2f + vorigin[0];
  float py = (iy + 0.5f) * 0.2f + vorigin[1];
  float pz = (iz + 0.5f) * 0.2f + vorigin[2];
  sp[(size_t)f * EMBED + c] = px * posw[c * 3 + 0] + py * posw[c * 3 + 1] + pz * posw[c * 3 + 2];
}

__global__ void k_full_init(const float* __restrict__ sew, const float* __restrict__ x3d,
                            _Float16* __restrict__ full) {
  size_t t = (size_t)blockIdx.x * 256 + threadIdx.x;
  if (t >= (size_t)NQ * EMBED) return;
  int c = (int)(t / NQ), q = (int)(t - (size_t)c * NQ);
  full[(size_t)q * EMBED + c] = (_Float16)(sew[(size_t)q * EMBED + c] + x3d[t]);
}

__global__ void k_scatter_se(const float* __restrict__ se, const int* __restrict__ fidx,
                             const int* __restrict__ cnts, _Float16* __restrict__ full) {
  int t = blockIdx.x * blockDim.x + threadIdx.x;
  int f = t >> 7, c = t & 127;
  if (f >= cnts[2]) return;
  full[(size_t)fidx[f] * EMBED + c] = (_Float16)se[(size_t)f * EMBED + c];
}

// ---------------- WMMA GEMM kernels ----------------
// out[M,N] = act( A @ W^T + bias ), A = (A1 (+A2)) f32, W is N x K.
// 128 threads = 4 waves; block tile 16(M) x 64(N); one 16x16 WMMA tile per wave.
__global__ void k_wmma_linear(const float* __restrict__ A1, const float* __restrict__ A2,
                              const float* __restrict__ W, const float* __restrict__ bias,
                              float* __restrict__ out, const int* __restrict__ m_cnt,
                              int Mstatic, int N, int K, int relu) {
  __shared__ _Float16 a_sh[16 * 512];
  const int tid = threadIdx.x;
  const int lane = tid & 31;
  const int wave = tid >> 5;
  const int M = m_cnt ? *m_cnt : Mstatic;
  const int mbase = blockIdx.y * 16;
  if (mbase >= M) return;
  for (int idx = tid; idx < 16 * K; idx += 128) {
    int r = idx / K, k = idx - r * K;
    int m = mbase + r;
    float v = 0.f;
    if (m < M) {
      v = A1[(size_t)m * K + k];
      if (A2) v += A2[(size_t)m * K + k];
    }
    a_sh[r * K + k] = (_Float16)v;
  }
  __syncthreads();
  const int nbase = blockIdx.x * 64 + wave * 16;
  if (nbase >= N) return;
  __builtin_prefetch(&W[(size_t)nbase * K], 0, 1);  // global_prefetch_b8
  v8f acc = {};
  for (int kb = 0; kb < K; kb += 32) {
    v16h a = fragA_lds(a_sh, K, kb, lane);
    v16h b = fragB_w(W, K, N, nbase, kb, lane);
    acc = wmma_f16(a, b, acc);
  }
  const int grp = lane >> 4;
  const int n = nbase + (lane & 15);
  if (n < N) {
#pragma unroll
    for (int r = 0; r < 8; ++r) {
      int m = mbase + r + grp * 8;
      if (m < M) {
        float v = acc[r] + (bias ? bias[n] : 0.f);
        if (relu) v = fmaxf(v, 0.f);
        out[(size_t)m * N + n] = v;
      }
    }
  }
}

// Fused FFN: y = x + relu(x @ W1^T + b1) @ W2^T + b2.  Hidden 16x512 tile lives in LDS.
__global__ void k_ffn_fused(const float* __restrict__ x,
                            const float* __restrict__ W1, const float* __restrict__ b1,
                            const float* __restrict__ W2, const float* __restrict__ b2,
                            float* __restrict__ y, const int* __restrict__ m_cnt) {
  __shared__ _Float16 x_sh[16 * EMBED];       // 4 KB
  __shared__ _Float16 h_sh[16 * 4 * EMBED];   // 16 KB
  const int tid = threadIdx.x, lane = tid & 31, wave = tid >> 5;
  const int M = *m_cnt;
  const int mbase = blockIdx.x * 16;
  if (mbase >= M) return;
  for (int idx = tid; idx < 16 * EMBED; idx += 128) {
    int r = idx >> 7, k = idx & 127;
    x_sh[idx] = (_Float16)((mbase + r < M) ? x[(size_t)(mbase + r) * EMBED + k] : 0.f);
  }
  __syncthreads();
  const int grp = lane >> 4;
  for (int tile = wave; tile < 32; tile += 4) {  // 512 hidden cols
    int nbase = tile * 16;
    v8f acc = {};
#pragma unroll
    for (int kb = 0; kb < EMBED; kb += 32) {
      v16h a = fragA_lds(x_sh, EMBED, kb, lane);
      v16h b = fragB_w(W1, EMBED, 512, nbase, kb, lane);
      acc = wmma_f16(a, b, acc);
    }
    int n = nbase + (lane & 15);
#pragma unroll
    for (int r = 0; r < 8; ++r) {
      int m = r + grp * 8;
      h_sh[m * 512 + n] = (_Float16)fmaxf(acc[r] + b1[n], 0.f);
    }
  }
  __syncthreads();
  for (int tile = wave; tile < 8; tile += 4) {   // 128 output cols
    int nbase = tile * 16;
    v8f acc = {};
#pragma unroll
    for (int kb = 0; kb < 512; kb += 32) {
      v16h a = fragA_lds(h_sh, 512, kb, lane);
      v16h b = fragB_w(W2, 512, EMBED, nbase, kb, lane);
      acc = wmma_f16(a, b, acc);
    }
    int n = nbase + (lane & 15);
#pragma unroll
    for (int r = 0; r < 8; ++r) {
      int m = mbase + r + grp * 8;
      if (m < M) y[(size_t)m * EMBED + n] = acc[r] + b2[n] + x[(size_t)m * EMBED + n];
    }
  }
}

// Final conv: out[o][q] = full[q][:] . convw[o][:] + convb[o]   (o < 20, q < NQ)
// Block stages a contiguous 64x128 f16 tile (16 KB) into LDS -- via the gfx1250
// async-to-LDS DMA path when the toolchain exposes it -- then all 4 waves run
// their own 16-row sub-tile over both 16-col n-tiles (8 unrolled WMMAs / wave).
__global__ void k_conv(const _Float16* __restrict__ full, const float* __restrict__ Wc,
                       const float* __restrict__ bc, float* __restrict__ out) {
  __shared__ _Float16 a_sh[64 * EMBED];   // 16 KB, contiguous image of 64 rows
  const int tid = threadIdx.x, lane = tid & 31, wave = tid >> 5;
  const int mblock = blockIdx.x * 64;
  const _Float16* gsrc = full + (size_t)mblock * EMBED;
#if __has_builtin(__builtin_amdgcn_global_load_async_to_lds_b128) && \
    __has_builtin(__builtin_amdgcn_s_wait_asynccnt)
  for (int i = tid; i < (64 * EMBED) / 8; i += 128) {   // 8 halves per B128 transfer
    __builtin_amdgcn_global_load_async_to_lds_b128(
        (b128_t*)(gsrc + i * 8),
        (b128_t*)(a_sh + i * 8), 0, 0);
  }
  __builtin_amdgcn_s_wait_asynccnt(0);
#else
  for (int i = tid; i < (64 * EMBED) / 8; i += 128) {
    ((int4*)a_sh)[i] = ((const int4*)gsrc)[i];
  }
#endif
  __syncthreads();
  const _Float16* aw = a_sh + wave * 16 * EMBED;  // this wave's 16 rows
  const int grp = lane >> 4;
#pragma unroll
  for (int nt = 0; nt < 2; ++nt) {                // n-tiles cover 0..31 (valid < 20)
    const int nbase = nt * 16;
    v8f acc = {};
#pragma unroll
    for (int kb = 0; kb < EMBED; kb += 32) {
      v16h a = fragA_lds(aw, EMBED, kb, lane);
      v16h b = fragB_w(Wc, EMBED, NCLS, nbase, kb, lane);
      acc = wmma_f16(a, b, acc);
    }
    const int n = nbase + (lane & 15);
    if (n < NCLS) {
#pragma unroll
      for (int r = 0; r < 8; ++r) {
        const int m = mblock + wave * 16 + r + grp * 8;
        out[(size_t)n * NQ + m] = acc[r] + bc[n];
      }
    }
  }
}

// ---------------- layernorm / small linear / attention ----------------

__global__ void k_ln(float* __restrict__ out, const float* __restrict__ in1,
                     const float* __restrict__ in2, const float* __restrict__ in3,
                     const float* __restrict__ g, const float* __restrict__ b,
                     const int* __restrict__ m_cnt) {
  __shared__ float sh[EMBED];
  int row = blockIdx.x;
  if (row >= *m_cnt) return;
  int t = threadIdx.x;
  float v = in1[(size_t)row * EMBED + t];
  if (in2) v += in2[(size_t)row * EMBED + t];
  if (in3) v += in3[(size_t)row * EMBED + t];
  sh[t] = v; __syncthreads();
  for (int st = 64; st > 0; st >>= 1) { if (t < st) sh[t] += sh[t + st]; __syncthreads(); }
  float mu = sh[0] * (1.f / EMBED);
  __syncthreads();
  float d = v - mu;
  sh[t] = d * d; __syncthreads();
  for (int st = 64; st > 0; st >>= 1) { if (t < st) sh[t] += sh[t + st]; __syncthreads(); }
  float var = sh[0] * (1.f / EMBED);
  out[(size_t)row * EMBED + t] = d * rsqrtf(var + 1e-5f) * g[t] + b[t];
}

__global__ void k_small_linear(const float* __restrict__ in1, const float* __restrict__ in2,
                               const float* __restrict__ W, const float* __restrict__ bias,
                               float* __restrict__ out, const int* __restrict__ cnt,
                               int N, int K, int relu) {
  int t = blockIdx.x * blockDim.x + threadIdx.x;
  int r = t / N, n = t - r * N;
  if (r >= *cnt) return;
  float acc = bias ? bias[n] : 0.f;
  for (int k = 0; k < K; ++k) {
    float a = in1[(size_t)r * K + k];
    if (in2) a += in2[(size_t)r * K + k];
    acc += a * W[(size_t)n * K + k];
  }
  if (relu) acc = fmaxf(acc, 0.f);
  out[(size_t)r * N + n] = acc;
}

// instance queries attend over F scene keys: online (chunked) softmax, 256 threads/block
__global__ void k_inst_attn(const float* __restrict__ qh, const float* __restrict__ kh,
                            const float* __restrict__ vh, float* __restrict__ o,
                            const int* __restrict__ cnts) {
  const int h = blockIdx.x, j = blockIdx.y;
  if (j >= cnts[1]) return;
  const int F = cnts[2];
  __shared__ float qv[HDIM];
  __shared__ float p_sh[256];
  __shared__ float red[256];
  const int t = threadIdx.x;
  if (t < HDIM) qv[t] = qh[(size_t)j * EMBED + h * HDIM + t];
  __syncthreads();
  const float sc = 0.17677669529663687f;  // 1/sqrt(32)
  float m = -INFINITY, s = 0.f, part = 0.f;
  const int kl = t >> 5, d = t & 31;
  for (int base = 0; base < F; base += 256) {
    int kk = base + t;
    float e = -INFINITY;
    if (kk < F) {
      float acc = 0.f;
      const float* kr = kh + (size_t)kk * EMBED + h * HDIM;
#pragma unroll
      for (int d2 = 0; d2 < HDIM; ++d2) acc += qv[d2] * kr[d2];
      e = acc * sc;
    }
    red[t] = e; __syncthreads();
    for (int st = 128; st > 0; st >>= 1) { if (t < st) red[t] = fmaxf(red[t], red[t + st]); __syncthreads(); }
    float mnew = fmaxf(m, red[0]);
    __syncthreads();
    float scale = (m == -INFINITY) ? 0.f : __expf(m - mnew);
    float p = (kk < F) ? __expf(e - mnew) : 0.f;
    p_sh[t] = p; red[t] = p; __syncthreads();
    for (int st = 128; st > 0; st >>= 1) { if (t < st) red[t] += red[t + st]; __syncthreads(); }
    s = s * scale + red[0];
    part *= scale;
    for (int ii = 0; ii < 32; ++ii) {
      int ck = (ii << 3) + kl;
      int kk2 = base + ck;
      if (kk2 < F) part += p_sh[ck] * vh[(size_t)kk2 * EMBED + h * HDIM + d];
    }
    m = mnew;
    __syncthreads();
  }
  red[t] = part; __syncthreads();
  if (kl == 0) {
    float tot = 0.f;
#pragma unroll
    for (int g2 = 0; g2 < 8; ++g2) tot += red[g2 * 32 + d];
    o[(size_t)j * EMBED + h * HDIM + d] = tot / s;
  }
}

// scene rows attend over <=NS_MAX instance keys: one thread per (row, head)
__global__ void k_scene_attn(const float* __restrict__ q, const float* __restrict__ kh,
                             const float* __restrict__ vh, float* __restrict__ o,
                             const int* __restrict__ cnts) {
  int t = blockIdx.x * blockDim.x + threadIdx.x;
  int f = t >> 2, h = t & 3;
  if (f >= cnts[2]) return;
  int nsel = cnts[1];
  float qv[HDIM];
  const float* qr = q + (size_t)f * EMBED + h * HDIM;
#pragma unroll
  for (int d = 0; d < HDIM; ++d) qv[d] = qr[d];
  const float sc = 0.17677669529663687f;
  float mx = -INFINITY;
  for (int jj = 0; jj < nsel; ++jj) {
    const float* kr = kh + (size_t)jj * EMBED + h * HDIM;
    float a = 0.f;
#pragma unroll
    for (int d = 0; d < HDIM; ++d) a += qv[d] * kr[d];
    mx = fmaxf(mx, a * sc);
  }
  float s = 0.f, acc[HDIM];
#pragma unroll
  for (int d = 0; d < HDIM; ++d) acc[d] = 0.f;
  for (int jj = 0; jj < nsel; ++jj) {
    const float* kr = kh + (size_t)jj * EMBED + h * HDIM;
    float a = 0.f;
#pragma unroll
    for (int d = 0; d < HDIM; ++d) a += qv[d] * kr[d];
    float e = __expf(a * sc - mx);
    s += e;
    const float* vr = vh + (size_t)jj * EMBED + h * HDIM;
#pragma unroll
    for (int d = 0; d < HDIM; ++d) acc[d] += e * vr[d];
  }
  float inv = (s > 0.f) ? 1.f / s : 0.f;
  float* orow = o + (size_t)f * EMBED + h * HDIM;
#pragma unroll
  for (int d = 0; d < HDIM; ++d) orow[d] = acc[d] * inv;
}

// ---------------- host launcher ----------------

extern "C" void kernel_launch(void* const* d_in, const int* in_sizes, int n_in,
                              void* d_out, int out_size, void* d_ws, size_t ws_size,
                              hipStream_t stream) {
  const float* queries = (const float*)d_in[0];
  const float* plogits = (const float*)d_in[1];
  const float* pmasks  = (const float*)d_in[2];
  const float* x3d     = (const float*)d_in[3];
  const float* depth   = (const float*)d_in[4];
  const float* Km      = (const float*)d_in[5];
  const float* Em      = (const float*)d_in[6];
  const float* vorigin = (const float*)d_in[7];
  const unsigned char* fovm = (const unsigned char*)d_in[8];
  const float* sew     = (const float*)d_in[9];
  const float* ipw     = (const float*)d_in[10];
  const float* posw    = (const float*)d_in[11];
  const float* convw   = (const float*)d_in[12];
  const float* convb   = (const float*)d_in[13];
  const float* attnw   = (const float*)d_in[14];
  const float* attnb   = (const float*)d_in[15];
  const float* lng     = (const float*)d_in[16];
  const float* lnbp    = (const float*)d_in[17];
  const float* fw1     = (const float*)d_in[18];
  const float* fb1     = (const float*)d_in[19];
  const float* fw2     = (const float*)d_in[20];
  const float* fb2     = (const float*)d_in[21];

  char* wsb = (char*)d_ws;
  size_t off = 0;
  auto alloc = [&](size_t bytes) -> void* {
    void* p = wsb + off;
    off = (off + bytes + 255) & ~(size_t)255;
    return p;
  };
  float* invm   = (float*)alloc(32 * 4);
  int*   cnts   = (int*)alloc(8 * 4);           // [0]=nk [1]=nsel [2]=F
  float* scores = (float*)alloc(NI * 4);
  int*   keepf  = (int*)alloc(NI * 4);
  int*   kidx   = (int*)alloc(NI * 4);
  int*   mids   = (int*)alloc((size_t)NPIX * 4);
  int*   selmap = (int*)alloc((size_t)NPIX * 4);
  int*   marea  = (int*)alloc(NI * 4);
  int*   parea  = (int*)alloc(NI * 4);
  int*   matm   = (int*)alloc(NI * 4);
  int*   remap  = (int*)alloc(NI * 4);
  int*   k2idx  = (int*)alloc(NS_MAX * 4);
  float* world  = (float*)alloc((size_t)NPIX * 3 * 4);
  float* xyz    = (float*)alloc(NS_MAX * 3 * 4);
  float* instq  = (float*)alloc((size_t)NS_MAX * EMBED * 4);
  float* instp  = (float*)alloc((size_t)NS_MAX * EMBED * 4);
  float* iq     = (float*)alloc((size_t)NS_MAX * EMBED * 4);
  float* xi     = (float*)alloc((size_t)NS_MAX * EMBED * 4);
  float* oi     = (float*)alloc((size_t)NS_MAX * EMBED * 4);
  float* ti     = (float*)alloc((size_t)NS_MAX * EMBED * 4);
  float* hi     = (float*)alloc((size_t)NS_MAX * 512 * 4);
  float* khs    = (float*)alloc((size_t)NS_MAX * EMBED * 4);
  float* vhs    = (float*)alloc((size_t)NS_MAX * EMBED * 4);
  int*   bcnt   = (int*)alloc(1024 * 4);
  int*   boff   = (int*)alloc(1024 * 4);
  int*   fidx   = (int*)alloc((size_t)F_MAX * 4);
  float* se     = (float*)alloc((size_t)F_MAX * EMBED * 4);
  float* sp     = (float*)alloc((size_t)F_MAX * EMBED * 4);
  float* bufA   = (float*)alloc((size_t)F_MAX * EMBED * 4);
  float* bufB   = (float*)alloc((size_t)F_MAX * EMBED * 4);
  float* bufC   = (float*)alloc((size_t)F_MAX * EMBED * 4);
  float* xs     = (float*)alloc((size_t)F_MAX * EMBED * 4);
  _Float16* fullh = (_Float16*)alloc((size_t)NQ * EMBED * 2);
  (void)in_sizes; (void)n_in; (void)out_size; (void)ws_size;

  auto AW  = [&](int i, int s, int j) { return attnw + (size_t)((i * 2 + s) * 4 + j) * EMBED * EMBED; };
  auto AB  = [&](int i, int s, int j) { return attnb + (size_t)((i * 2 + s) * 4 + j) * EMBED; };
  auto LG  = [&](int i, int s, int k) { return lng  + (size_t)((i * 2 + s) * 2 + k) * EMBED; };
  auto LBt = [&](int i, int s, int k) { return lnbp + (size_t)((i * 2 + s) * 2 + k) * EMBED; };
  auto F1W = [&](int i, int s) { return fw1 + (size_t)(i * 2 + s) * 512 * EMBED; };
  auto F1B = [&](int i, int s) { return fb1 + (size_t)(i * 2 + s) * 512; };
  auto F2W = [&](int i, int s) { return fw2 + (size_t)(i * 2 + s) * EMBED * 512; };
  auto F2B = [&](int i, int s) { return fb2 + (size_t)(i * 2 + s) * EMBED; };

  const float* FNULL = nullptr;

  // ---- selection + geometry ----
  k_invmats<<<1, 1, 0, stream>>>(Km, Em, invm);
  k_scores<<<2, 256, 0, stream>>>(plogits, scores, keepf);
  k_compact_kidx<<<1, 1, 0, stream>>>(keepf, kidx, cnts);
  k_mids<<<NPIX / 256, 256, 0, stream>>>(pmasks, scores, kidx, cnts, mids);
  k_zero_i32<<<2, 256, 0, stream>>>(marea, NI);
  k_zero_i32<<<2, 256, 0, stream>>>(parea, NI);
  k_zero_i32<<<2, 256, 0, stream>>>(matm, NI);
  k_zero_f32<<<1, 256, 0, stream>>>(xyz, NS_MAX * 3);
  k_areas<<<NPIX / 256, 256, 0, stream>>>(pmasks, kidx, cnts, mids, marea, parea, matm, selmap);
  k_compact_sel<<<1, 1, 0, stream>>>(cnts, kidx, marea, parea, matm, remap, k2idx, cnts);
  k_world<<<NPIX / 256, 256, 0, stream>>>(depth, invm, world);
  k_xyz<<<NPIX / 256, 256, 0, stream>>>(selmap, remap, world, xyz);
  k_instqp<<<NS_MAX * EMBED / 256, 256, 0, stream>>>(queries, ipw, posw, k2idx, xyz, cnts, instq, instp);

  // ---- fov compaction + scene embeddings ----
  k_fov_count<<<1024, 256, 0, stream>>>(fovm, bcnt);
  k_fov_scan<<<1, 1024, 0, stream>>>(bcnt, boff, cnts);
  k_fov_scatter<<<1024, 256, 0, stream>>>(fovm, boff, fidx);
  k_se_sp<<<F_MAX * EMBED / 256, 256, 0, stream>>>(sew, x3d, posw, vorigin, fidx, cnts, se, sp);
  k_full_init<<<(NQ / 256) * EMBED, 256, 0, stream>>>(sew, x3d, fullh);

  const dim3 gl(2, F_MAX / 16);  // N=128 -> 2 blocks of 64 cols; 16-row tiles over F
  for (int i = 0; i < NLAYERS; ++i) {
    // ---- instance-side decoder layer ----
    k_small_linear<<<NS_MAX * EMBED / 256, 256, 0, stream>>>(instq, instp, AW(i, 0, 0), AB(i, 0, 0), iq, cnts + 1, EMBED, EMBED, 0);
    k_wmma_linear<<<gl, 128, 0, stream>>>(se, sp, AW(i, 0, 1), AB(i, 0, 1), bufA, cnts + 2, 0, EMBED, EMBED, 0);
    k_wmma_linear<<<gl, 128, 0, stream>>>(se, FNULL, AW(i, 0, 2), AB(i, 0, 2), bufB, cnts + 2, 0, EMBED, EMBED, 0);
    k_inst_attn<<<dim3(HEADS, NS_MAX), 256, 0, stream>>>(iq, bufA, bufB, oi, cnts);
    k_small_linear<<<NS_MAX * EMBED / 256, 256, 0, stream>>>(oi, FNULL, AW(i, 0, 3), AB(i, 0, 3), ti, cnts + 1, EMBED, EMBED, 0);
    k_ln<<<NS_MAX, EMBED, 0, stream>>>(xi, instq, instp, ti, LG(i, 0, 0), LBt(i, 0, 0), cnts + 1);
    k_small_linear<<<NS_MAX * 512 / 256, 256, 0, stream>>>(xi, FNULL, F1W(i, 0), F1B(i, 0), hi, cnts + 1, 512, EMBED, 1);
    k_small_linear<<<NS_MAX * EMBED / 256, 256, 0, stream>>>(hi, FNULL, F2W(i, 0), F2B(i, 0), ti, cnts + 1, EMBED, 512, 0);
    k_ln<<<NS_MAX, EMBED, 0, stream>>>(instq, xi, ti, FNULL, LG(i, 0, 1), LBt(i, 0, 1), cnts + 1);
    // ---- scene-side decoder layer ----
    k_small_linear<<<NS_MAX * EMBED / 256, 256, 0, stream>>>(instq, instp, AW(i, 1, 1), AB(i, 1, 1), khs, cnts + 1, EMBED, EMBED, 0);
    k_small_linear<<<NS_MAX * EMBED / 256, 256, 0, stream>>>(instq, FNULL, AW(i, 1, 2), AB(i, 1, 2), vhs, cnts + 1, EMBED, EMBED, 0);
    k_wmma_linear<<<gl, 128, 0, stream>>>(se, sp, AW(i, 1, 0), AB(i, 1, 0), bufA, cnts + 2, 0, EMBED, EMBED, 0);
    k_scene_attn<<<F_MAX * HEADS / 256, 256, 0, stream>>>(bufA, khs, vhs, bufB, cnts);
    k_wmma_linear<<<gl, 128, 0, stream>>>(bufB, FNULL, AW(i, 1, 3), AB(i, 1, 3), bufC, cnts + 2, 0, EMBED, EMBED, 0);
    k_ln<<<F_MAX, EMBED, 0, stream>>>(xs, se, sp, bufC, LG(i, 1, 0), LBt(i, 1, 0), cnts + 2);
    k_ffn_fused<<<F_MAX / 16, 128, 0, stream>>>(xs, F1W(i, 1), F1B(i, 1), F2W(i, 1), F2B(i, 1), bufC, cnts + 2);
    k_ln<<<F_MAX, EMBED, 0, stream>>>(se, bufC, FNULL, FNULL, LG(i, 1, 1), LBt(i, 1, 1), cnts + 2);
  }

  // ---- scatter refined scene rows + final 128->20 projection over all voxels ----
  k_scatter_se<<<F_MAX * EMBED / 256, 256, 0, stream>>>(se, fidx, cnts, fullh);
  k_conv<<<NQ / 64, 128, 0, stream>>>(fullh, convw, convb, (float*)d_out);
}
